// SelfAttention_83150566851162
// MI455X (gfx1250) — matrix operand order repaired and checked by reading, not compiled, not run on hardware
//
#include <hip/hip_runtime.h>
#include <hip/hip_bf16.h>

// ---------------------------------------------------------------------------
// Self-attention (BS=8, N=2048, D=512, fp32 I/O) for MI455X (gfx1250).
//  - bf16 WMMA (v_wmma_f32_16x16x32_bf16) for all GEMMs, fp32 accumulate.
//  - Scores computed TRANSPOSED (S' = K*Q^T): online softmax is lane-local
//    + one shfl_xor(16); exp(S') lands directly in the WMMA A-operand layout.
//  - QKV: 4-wave blocks; W e-tiles shared via LDS, double-buffered with
//    GLOBAL_LOAD_ASYNC_TO_LDS_B128 (ASYNCcnt) -> overlap + 4x less L2 traffic.
//  - Flash: K tiles staged in LDS per block (shared by 4 waves), also
//    double-buffered with async loads.
//  - LDS rows padded to 520 bf16 so 16-lane row-strided ds_load_b128 are
//    bank-conflict-free (1040B row stride -> bank stride 4 dwords).
// ---------------------------------------------------------------------------

#define BSZ   8
#define NSEQ  2048
#define DM    512
#define LPAD  520            // padded LDS row stride in bf16 elements

typedef float  v8f   __attribute__((ext_vector_type(8)));
typedef float  v4f   __attribute__((ext_vector_type(4)));
typedef int    v4i   __attribute__((ext_vector_type(4)));
typedef __bf16 v16bf __attribute__((ext_vector_type(16)));
typedef __bf16 v8bf  __attribute__((ext_vector_type(8)));
typedef __bf16 v4bf  __attribute__((ext_vector_type(4)));

#define GAS __attribute__((address_space(1)))
#define LAS __attribute__((address_space(3)))

#if __has_builtin(__builtin_amdgcn_global_load_async_to_lds_b128) && \
    __has_builtin(__builtin_amdgcn_s_wait_asynccnt)
#define HAVE_ASYNC_LDS 1
#else
#define HAVE_ASYNC_LDS 0
#endif

__device__ __forceinline__ v8bf ld8(const __bf16* p) {
    return *(const v8bf*)p;
}

__device__ __forceinline__ v16bf cat16(v8bf lo, v8bf hi) {
    union { v16bf w; v8bf h[2]; } u;
    u.h[0] = lo; u.h[1] = hi;
    return u.w;
}

__device__ __forceinline__ v8f wmma_bf16(v16bf a, v16bf b, v8f c) {
    // D = A(16x32 bf16) * B(32x16 bf16) + C(16x16 f32)
    return __builtin_amdgcn_wmma_f32_16x16x32_bf16(
        false, a, false, b, (short)0, c, false, false);
}

// ---------------------------------------------------------------------------
// fp32 -> bf16 conversion (vectorized x4)
// ---------------------------------------------------------------------------
__global__ void cvt_f32_bf16(const float* __restrict__ src,
                             __bf16* __restrict__ dst, int n4) {
    int i = blockIdx.x * blockDim.x + threadIdx.x;
    int stride = gridDim.x * blockDim.x;
    const v4f* s4 = (const v4f*)src;
    v4bf* d4 = (v4bf*)dst;
    for (; i < n4; i += stride) {
        v4f x = s4[i];
        v4bf y;
        y[0] = (__bf16)x[0]; y[1] = (__bf16)x[1];
        y[2] = (__bf16)x[2]; y[3] = (__bf16)x[3];
        d4[i] = y;
    }
}

// ---------------------------------------------------------------------------
// QKV projection: block = 4 waves, each wave owns one 16-row tile of H
// (A operand resident in 128 VGPRs). The 16x512 W e-tile is shared by all
// waves through LDS and double-buffered with async loads.
// gridDim.x = 3 * (BSZ*NSEQ/16)/4 = 768 blocks of 128 threads.
// ---------------------------------------------------------------------------
__global__ __launch_bounds__(128)
void qkv_kernel(const __bf16* __restrict__ Hb,
                const __bf16* __restrict__ Wqb,
                const __bf16* __restrict__ Wkb,
                const __bf16* __restrict__ Wvb,
                const float* __restrict__ bq,
                const float* __restrict__ bk,
                const float* __restrict__ bv,
                __bf16* __restrict__ Qb,
                __bf16* __restrict__ Kb,
                __bf16* __restrict__ Vtb) {
    const int mat  = blockIdx.x >> 8;         // 0=Q, 1=K, 2=V (256 blocks each)
    const int g    = blockIdx.x & 255;        // group of 4 row-tiles
    const int tid  = threadIdx.x;
    const int wave = tid >> 5;
    const int lane = tid & 31;
    const int l15  = lane & 15;
    const int hi   = lane >> 4;
    const int row0 = (g * 4 + wave) * 16;     // this wave's 16-row tile

    const __bf16* W    = (mat == 0) ? Wqb : (mat == 1 ? Wkb : Wvb);
    const float*  bias = (mat == 0) ? bq  : (mat == 1 ? bk  : bv);
    const float scale  = (mat == 0) ? 0.04419417382415922f : 1.0f;

    __shared__ __bf16 wsm[2][16 * LPAD];      // 2 x ~16.6 KB W e-tiles

    // A (16 rows x 512 of Hb) resident in registers, A-operand layout
    v16bf A[16];
    const __bf16* hrow = Hb + (size_t)(row0 + l15) * DM;
    #pragma unroll
    for (int ds = 0; ds < 16; ++ds)
        A[ds] = cat16(ld8(hrow + ds * 32 + hi * 16),
                      ld8(hrow + ds * 32 + hi * 16 + 8));

#if HAVE_ASYNC_LDS
    // one W e-tile = 16 rows x 64 b128 = 1024 b128; 8 per thread; ASYNCcnt += 8
    #define W_ISSUE(buf, et)                                                   \
        {                                                                      \
            const __bf16* gsrc = W + (size_t)(et) * 16 * DM;                   \
            _Pragma("unroll") for (int j = 0; j < 8; ++j) {                    \
                int idx = j * 128 + tid;                                       \
                int row = idx >> 6, col = idx & 63;                            \
                __builtin_amdgcn_global_load_async_to_lds_b128(                \
                    (GAS v4i*)(gsrc + (size_t)row * DM + col * 8),             \
                    (LAS v4i*)(wsm[buf] + row * LPAD + col * 8),               \
                    0, 0);                                                     \
            }                                                                  \
        }
    W_ISSUE(0, 0)
#else
    #define W_COPY(buf, et)                                                    \
        {                                                                      \
            const __bf16* gsrc = W + (size_t)(et) * 16 * DM;                   \
            _Pragma("unroll") for (int j = 0; j < 8; ++j) {                    \
                int idx = j * 128 + tid;                                       \
                int row = idx >> 6, col = idx & 63;                            \
                *(uint4*)(wsm[buf] + row * LPAD + col * 8) =                   \
                    *(const uint4*)(gsrc + (size_t)row * DM + col * 8);        \
            }                                                                  \
        }
#endif

    const int b  = row0 >> 11;
    const int n0 = row0 & (NSEQ - 1);

    for (int et = 0; et < 32; ++et) {
        const int cur = et & 1;
#if HAVE_ASYNC_LDS
        if (et + 1 < 32) {
            W_ISSUE(cur ^ 1, et + 1)                  // prefetch next e-tile
            __builtin_amdgcn_s_wait_asynccnt(8);       // current tile landed
        } else {
            __builtin_amdgcn_s_wait_asynccnt(0);
        }
        __syncthreads();
        const __bf16* wrow = wsm[cur] + l15 * LPAD + hi * 16;
#else
        W_COPY(0, et)
        __syncthreads();
        const __bf16* wrow = wsm[0] + l15 * LPAD + hi * 16;
#endif

        v8f c;
        #pragma unroll
        for (int r = 0; r < 8; ++r) c[r] = 0.0f;

        #pragma unroll
        for (int ds = 0; ds < 16; ++ds) {
            v16bf B = cat16(ld8(wrow + ds * 32), ld8(wrow + ds * 32 + 8));
            c = wmma_bf16(A[ds], B, c);
        }

        const int e = et * 16 + l15;
        const float bval = bias[e];
        #pragma unroll
        for (int r = 0; r < 8; ++r) c[r] = (c[r] + bval) * scale;

        if (mat < 2) {
            __bf16* dst = (mat == 0) ? Qb : Kb;
            const int base = hi * 8;
            #pragma unroll
            for (int r = 0; r < 8; ++r)
                dst[(size_t)(row0 + base + r) * DM + e] = (__bf16)c[r];
        } else {
            v8bf pk;
            #pragma unroll
            for (int r = 0; r < 8; ++r) pk[r] = (__bf16)c[r];
            *(v8bf*)(Vtb + ((size_t)b * DM + e) * NSEQ + n0 + hi * 8) = pk;
        }
        __syncthreads();   // all waves done with wsm[cur] before refill
    }
}

// ---------------------------------------------------------------------------
// Flash attention: block = 4 waves, one 16-query tile per block.
// Q tile + double-buffered 32-key K tile live in LDS (padded rows).
// ---------------------------------------------------------------------------
__global__ __launch_bounds__(128)
void flash_kernel(const __bf16* __restrict__ Qb,
                  const __bf16* __restrict__ Kb,
                  const __bf16* __restrict__ Vtb,
                  float* __restrict__ out) {
    const int b    = blockIdx.x >> 7;
    const int qt   = blockIdx.x & 127;
    const int tid  = threadIdx.x;
    const int wave = tid >> 5;
    const int lane = tid & 31;
    const int l15  = lane & 15;
    const int hi   = lane >> 4;
    const int base = hi * 8;
    const int dbase = wave * 128;

    __shared__ __bf16 qs[16 * LPAD];          // ~16.6 KB query tile (padded)
    __shared__ __bf16 ks[2][32 * LPAD];       // 2 x ~33.3 KB K tiles (padded)

    const __bf16* Kbase = Kb  + (size_t)b * NSEQ * DM;
    const __bf16* Vbase = Vtb + (size_t)b * DM * NSEQ;

    // ---- cooperative Q tile copy (row-padded) ----
    {
        const uint4* src = (const uint4*)(Qb + ((size_t)b * NSEQ + qt * 16) * DM);
        #pragma unroll
        for (int j = 0; j < 8; ++j) {
            int idx = j * 128 + tid;          // 0..1023
            int row = idx >> 6, col = idx & 63;
            *(uint4*)(qs + row * LPAD + col * 8) = src[row * (DM / 8) + col];
        }
    }

#if HAVE_ASYNC_LDS
    // async K tile fill: 2048 x b128 per buffer, 16 per thread, ASYNCcnt += 16
    #define K_ISSUE(buf, kb)                                                   \
        {                                                                      \
            const __bf16* gsrc = Kbase + (size_t)(kb)*DM;                      \
            _Pragma("unroll") for (int j = 0; j < 16; ++j) {                   \
                int idx = j * 128 + tid;                                       \
                int row = idx >> 6, col = idx & 63;                            \
                __builtin_amdgcn_global_load_async_to_lds_b128(                \
                    (GAS v4i*)(gsrc + (size_t)row * DM + col * 8),             \
                    (LAS v4i*)(ks[buf] + row * LPAD + col * 8),                \
                    0, 0);                                                     \
            }                                                                  \
        }
    K_ISSUE(0, 0)
#else
    #define K_COPY(buf, kb)                                                    \
        {                                                                      \
            const __bf16* gsrc = Kbase + (size_t)(kb)*DM;                      \
            _Pragma("unroll") for (int j = 0; j < 16; ++j) {                   \
                int idx = j * 128 + tid;                                       \
                int row = idx >> 6, col = idx & 63;                            \
                *(uint4*)(ks[buf] + row * LPAD + col * 8) =                    \
                    *(const uint4*)(gsrc + (size_t)row * DM + col * 8);        \
            }                                                                  \
        }
#endif

    v8f O[8];
    #pragma unroll
    for (int t = 0; t < 8; ++t)
        #pragma unroll
        for (int r = 0; r < 8; ++r) O[t][r] = 0.0f;

    float m    = -__builtin_inff();
    float lsum = 0.0f;

    for (int i = 0; i < NSEQ / 32; ++i) {
        const int kb  = i * 32;
        const int cur = i & 1;

#if HAVE_ASYNC_LDS
        if (i + 1 < NSEQ / 32) {
            K_ISSUE((i + 1) & 1, kb + 32)             // prefetch next buffer
            __builtin_amdgcn_s_wait_asynccnt(16);      // current buffer landed
        } else {
            __builtin_amdgcn_s_wait_asynccnt(0);
        }
        __syncthreads();
        const __bf16* kbuf = ks[cur];
#else
        K_COPY(0, kb)
        __syncthreads();
        const __bf16* kbuf = ks[0];
#endif

        // ---- S' = K(32x512) * Q^T(512x16), two 16x16 f32 tiles, from LDS ----
        v8f S0, S1;
        #pragma unroll
        for (int r = 0; r < 8; ++r) { S0[r] = 0.0f; S1[r] = 0.0f; }

        const __bf16* krow0 = kbuf + l15 * LPAD;
        const __bf16* krow1 = krow0 + 16 * LPAD;
        const __bf16* qrow  = qs + l15 * LPAD + hi * 8;

        #pragma unroll
        for (int ds = 0; ds < 16; ++ds) {
            v16bf A0 = cat16(ld8(krow0 + ds * 32 + hi * 8),
                             ld8(krow0 + ds * 32 + 16 + hi * 8));
            v16bf A1 = cat16(ld8(krow1 + ds * 32 + hi * 8),
                             ld8(krow1 + ds * 32 + 16 + hi * 8));
            v16bf Bq = cat16(ld8(qrow + ds * 32), ld8(qrow + ds * 32 + 16));
            S0 = wmma_bf16(A0, Bq, S0);
            S1 = wmma_bf16(A1, Bq, S1);
        }

        // ---- online softmax: lane holds 16 key-scores of query q=l15 ----
        float mloc = S0[0];
        #pragma unroll
        for (int r = 0; r < 8; ++r) {
            mloc = fmaxf(mloc, S0[r]);
            mloc = fmaxf(mloc, S1[r]);
        }
        mloc = fmaxf(mloc, __shfl_xor(mloc, 16));
        const float mnew  = fmaxf(m, mloc);
        const float alpha = __expf(m - mnew);

        float P0[8], P1[8];
        float ssum = 0.0f;
        #pragma unroll
        for (int r = 0; r < 8; ++r) {
            P0[r] = __expf(S0[r] - mnew);
            P1[r] = __expf(S1[r] - mnew);
            ssum += P0[r] + P1[r];
        }
        ssum += __shfl_xor(ssum, 16);
        lsum  = lsum * alpha + ssum;
        m     = mnew;

        float av[8];
        #pragma unroll
        for (int r = 0; r < 8; ++r) av[r] = __shfl(alpha, base + r);
        #pragma unroll
        for (int t = 0; t < 8; ++t)
            #pragma unroll
            for (int r = 0; r < 8; ++r) O[t][r] *= av[r];

        // pack P into WMMA A-operand layout (lane-local by construction)
        v16bf P;
        #pragma unroll
        for (int r = 0; r < 8; ++r) {
            P[r]     = (__bf16)P0[r];
            P[8 + r] = (__bf16)P1[r];
        }

        // ---- O += P(16x32) * V(32 x d-chunk) from transposed V ----
        #pragma unroll
        for (int t = 0; t < 8; ++t) {
            const __bf16* vrow =
                Vbase + (size_t)(dbase + t * 16 + l15) * NSEQ + kb + hi * 8;
            v16bf Bv = cat16(ld8(vrow), ld8(vrow + 16));
            O[t] = wmma_bf16(P, Bv, O[t]);
        }

        __syncthreads();   // all waves done with kbuf before it is refilled
    }

    // ---- normalize and store fp32 output ----
    const float inv = 1.0f / lsum;
    float iv[8];
    #pragma unroll
    for (int r = 0; r < 8; ++r) iv[r] = __shfl(inv, base + r);

    const size_t rowb = (size_t)b * NSEQ + qt * 16;
    #pragma unroll
    for (int t = 0; t < 8; ++t)
        #pragma unroll
        for (int r = 0; r < 8; ++r)
            out[(rowb + base + r) * DM + dbase + t * 16 + l15] = O[t][r] * iv[r];
}

// ---------------------------------------------------------------------------
// Launch. Workspace: Hb(16M) Wq/Wk/Wv(0.5M x3) Qb(16M) Kb(16M) Vtb(16M) bytes.
// ---------------------------------------------------------------------------
extern "C" void kernel_launch(void* const* d_in, const int* in_sizes, int n_in,
                              void* d_out, int out_size, void* d_ws, size_t ws_size,
                              hipStream_t stream) {
    const float* H  = (const float*)d_in[0];
    const float* Wq = (const float*)d_in[1];
    const float* bq = (const float*)d_in[2];
    const float* Wk = (const float*)d_in[3];
    const float* bk = (const float*)d_in[4];
    const float* Wv = (const float*)d_in[5];
    const float* bv = (const float*)d_in[6];
    float* out = (float*)d_out;

    const size_t HB = (size_t)BSZ * NSEQ * DM * 2;
    const size_t WB = (size_t)DM * DM * 2;

    char* w = (char*)d_ws;
    __bf16* Hb  = (__bf16*)w;            w += HB;
    __bf16* Wqb = (__bf16*)w;            w += WB;
    __bf16* Wkb = (__bf16*)w;            w += WB;
    __bf16* Wvb = (__bf16*)w;            w += WB;
    __bf16* Qb  = (__bf16*)w;            w += HB;
    __bf16* Kb  = (__bf16*)w;            w += HB;
    __bf16* Vtb = (__bf16*)w;            w += HB;

    cvt_f32_bf16<<<2048, 256, 0, stream>>>(H,  Hb,  (BSZ * NSEQ * DM) / 4);
    cvt_f32_bf16<<<256,  256, 0, stream>>>(Wq, Wqb, (DM * DM) / 4);
    cvt_f32_bf16<<<256,  256, 0, stream>>>(Wk, Wkb, (DM * DM) / 4);
    cvt_f32_bf16<<<256,  256, 0, stream>>>(Wv, Wvb, (DM * DM) / 4);

    // 4 row-tiles per block share W e-tiles through LDS
    qkv_kernel<<<3 * (BSZ * NSEQ / 16) / 4, 128, 0, stream>>>(
        Hb, Wqb, Wkb, Wvb, bq, bk, bv, Qb, Kb, Vtb);

    flash_kernel<<<BSZ * (NSEQ / 16), 128, 0, stream>>>(Qb, Kb, Vtb, out);
}
